// WeightedAverageAttention_47545287967535
// MI455X (gfx1250) — compile-verified, matrix operand order checked by reading, not run on hardware
//
#include <hip/hip_runtime.h>

typedef __bf16 bf16_t;
typedef __attribute__((ext_vector_type(16))) __bf16 v16bf;
typedef __attribute__((ext_vector_type(8)))  __bf16 v8bf;
typedef __attribute__((ext_vector_type(8)))  float   v8f;
typedef __attribute__((ext_vector_type(4)))  float   v4f;
typedef __attribute__((ext_vector_type(4)))  unsigned int u32x4;
typedef __attribute__((ext_vector_type(8)))  int     i32x8;
typedef __attribute__((ext_vector_type(4)))  int     i32x4;

#define WMMA_BF16(a,b,c) __builtin_amdgcn_wmma_f32_16x16x32_bf16( \
    false, (a), false, (b), (short)0, (c), false, false)

constexpr int BB = 4, S = 4096, D = 256, TWO_D = 512;
constexpr int MROWS = BB * S;          // 16384 total rows
constexpr int KV_TILE = 32;            // kv rows staged per iteration
constexpr int NSLICE = 4;              // split-KV factor (occupancy)
constexpr int SLICE_ROWS = S / NSLICE; // 1024 kv rows per block
constexpr int NIT = SLICE_ROWS / KV_TILE;

#if __has_builtin(__builtin_amdgcn_tensor_load_to_lds) && \
    __has_builtin(__builtin_amdgcn_s_wait_tensorcnt)
#define USE_TDM 1
#endif

// ---------------------------------------------------------------------------
// Kernel 1: fp32 -> bf16 cast (x -> xb serves as V and proj A; W -> Wb)
// ---------------------------------------------------------------------------
__global__ void __launch_bounds__(256)
cast_kernel(const float* __restrict__ src, bf16_t* __restrict__ dst) {
    const size_t idx = ((size_t)blockIdx.x * blockDim.x + threadIdx.x) * 8;
    v4f a = *(const v4f*)(src + idx);
    v4f b = *(const v4f*)(src + idx + 4);
    v8bf o;
#pragma unroll
    for (int u = 0; u < 4; ++u) { o[u] = (__bf16)a[u]; o[u + 4] = (__bf16)b[u]; }
    *(v8bf*)(dst + idx) = o;
}

// ---------------------------------------------------------------------------
// Kernel 2: qk projection GEMM on pre-cast bf16 (qk = x@W^T + b)
// M=16384, N=512, K=256.  Block: 8 waves = 4(M) x 2(N); wave tile 16x64.
// ---------------------------------------------------------------------------
__global__ void __launch_bounds__(256)
proj_kernel(const bf16_t* __restrict__ xb, const bf16_t* __restrict__ Wb,
            const float* __restrict__ bias,
            bf16_t* __restrict__ qb, bf16_t* __restrict__ kb) {
    const int lane  = threadIdx.x & 31;
    const int wid   = threadIdx.x >> 5;
    const int laneN = lane & 15, half = lane >> 4;
    const int warpM = wid & 3,   warpN = wid >> 2;
    const int m0 = blockIdx.x * 64  + warpM * 16;
    const int n0 = blockIdx.y * 128 + warpN * 64;

    v8f c[4] = {};
#pragma unroll
    for (int ks = 0; ks < 8; ++ks) {
        const int k = ks * 32 + half * 16;
        v16bf a = *(const v16bf*)(xb + (size_t)(m0 + laneN) * D + k);
#pragma unroll
        for (int nt = 0; nt < 4; ++nt) {
            v16bf bm = *(const v16bf*)(Wb + (size_t)(n0 + nt * 16 + laneN) * D + k);
            c[nt] = WMMA_BF16(a, bm, c[nt]);
        }
    }
#pragma unroll
    for (int nt = 0; nt < 4; ++nt) {
        const int e0 = n0 + nt * 16;
        const float bval = bias[e0 + laneN];
        bf16_t* dst = (e0 < D) ? (qb + e0) : (kb + (e0 - D));
#pragma unroll
        for (int i = 0; i < 8; ++i) {
            const int r = m0 + i + half * 8;            // C-layout row
            dst[(size_t)r * D + laneN] = (__bf16)(c[nt][i] + bval);
        }
    }
}

// ---------------------------------------------------------------------------
// TDM: DMA one contiguous 32x256 bf16 K tile (16 KB) from global into LDS.
// D# layout per CDNA5 ISA 8.3/8.4 (count=1, type=2, data_size=2B, 1D tile).
// ---------------------------------------------------------------------------
#ifdef USE_TDM
__device__ __forceinline__ void tdm_load_tile(const bf16_t* gsrc, unsigned lds_off) {
    const unsigned long long ga = (unsigned long long)(uintptr_t)gsrc;
    u32x4 g0;
    g0[0] = 1u;                                          // count=1, user desc
    g0[1] = lds_off;                                     // lds_addr (bytes)
    g0[2] = (unsigned)(ga & 0xFFFFFFFFu);                // global_addr[31:0]
    g0[3] = (unsigned)((ga >> 32) & 0x01FFFFFFu) | (2u << 30); // addr hi, type=2
    i32x8 g1;
    g1[0] = 1 << 16;                                     // data_size=1 (2 bytes)
    g1[1] = (int)(8192u << 16);                          // tensor_dim0[15:0]=8192
    g1[2] = 1 << 16;                                     // tensor_dim1=1
    g1[3] = (int)(8192u << 16);                          // tile_dim0=8192 elems
    g1[4] = 1;                                           // tile_dim1=1
    g1[5] = 8192;                                        // tensor_dim0_stride
    g1[6] = 0; g1[7] = 0;
    i32x4 gz = {};
#if defined(__clang_major__) && (__clang_major__ >= 23)
    i32x8 gz8 = {};
    __builtin_amdgcn_tensor_load_to_lds(g0, g1, gz, gz, gz8, 0);
#else
    __builtin_amdgcn_tensor_load_to_lds(g0, g1, gz, gz, 0);
#endif
}
#endif

// ---------------------------------------------------------------------------
// Kernel 3: split-KV flash attention, double-buffered KV pipeline.
// grid=(S/128, B, NSLICE), 8 waves/block; each wave owns 16 query rows.
// Per iteration: issue TDM(K[it+1]) + V[it+1] loads into alt buffers, compute
// on current buffers, drain TENSORcnt, single barrier.  Emits unnormalized O
// (fp32) + per-row (m,l) for the merge kernel.
// ---------------------------------------------------------------------------
__global__ void __launch_bounds__(256)
attn_kernel(const bf16_t* __restrict__ qb, const bf16_t* __restrict__ kb,
            const bf16_t* __restrict__ xb, float* __restrict__ Opart,
            float* __restrict__ Mpart, float* __restrict__ Lpart) {
    __shared__ bf16_t Ks[2][KV_TILE * 256];   // [kv][d]   2 x 16 KB
    __shared__ bf16_t Vt[2][256 * KV_TILE];   // [d][kv]   2 x 16 KB (transposed)
    __shared__ bf16_t Ps[8][16 * KV_TILE];    // per-wave P staging, 8 KB

    const int tid   = threadIdx.x;
    const int lane  = tid & 31, wid = tid >> 5;
    const int laneN = lane & 15, half = lane >> 4;
    const int batch = blockIdx.y;
    const int slice = blockIdx.z;
    const int q0    = blockIdx.x * 128 + wid * 16;
    const size_t mbase = (size_t)(batch * S + q0) * D;

    // Resident Q fragments: 16 rows x 256 d (bf16) = 64 VGPRs
    v16bf aq[8];
#pragma unroll
    for (int ks = 0; ks < 8; ++ks)
        aq[ks] = *(const v16bf*)(qb + mbase + (size_t)laneN * D + ks * 32 + half * 16);

    v8f o[16] = {};
    float mrow[8], lrow[8];
#pragma unroll
    for (int i = 0; i < 8; ++i) { mrow[i] = -3.0e38f; lrow[i] = 0.0f; }

    const bf16_t* kbb = kb + (size_t)batch * S * D;
    const bf16_t* vbb = xb + (size_t)batch * S * D;
    const size_t tb0  = (size_t)slice * SLICE_ROWS * D;
    const float scale = 0.0625f;   // 1/sqrt(256)

    // V tile loader: coalesced global read, transposed scatter into LDS buf
    auto load_v_tile = [&](int buf, size_t tbase) {
#pragma unroll
        for (int j = 0; j < 4; ++j) {
            const int e = tid * 8 + j * 2048;
            const int row = e >> 8, col = e & 255;
            v8bf vv = *(const v8bf*)(vbb + tbase + e);
#pragma unroll
            for (int u = 0; u < 8; ++u) Vt[buf][(col + u) * KV_TILE + row] = vv[u];
        }
    };
#ifndef USE_TDM
    auto load_k_tile = [&](int buf, size_t tbase) {
#pragma unroll
        for (int j = 0; j < 4; ++j) {
            const int e = tid * 8 + j * 2048;
            *(v8bf*)(&Ks[buf][e]) = *(const v8bf*)(kbb + tbase + e);
        }
    };
#endif

    // ---- prologue: fill buffer 0
#ifdef USE_TDM
    if (wid == 0) tdm_load_tile(kbb + tb0, (unsigned)(uintptr_t)&Ks[0][0]);
#else
    load_k_tile(0, tb0);
#endif
    load_v_tile(0, tb0);
#ifdef USE_TDM
    if (wid == 0) __builtin_amdgcn_s_wait_tensorcnt(0);
#endif
    __syncthreads();

    for (int it = 0; it < NIT; ++it) {
        const int cur = it & 1, nxt = cur ^ 1;

        // ---- stage next tile into the alternate buffers (overlaps compute)
        if (it + 1 < NIT) {
            const size_t tnext = tb0 + (size_t)(it + 1) * KV_TILE * D;
#ifdef USE_TDM
            if (wid == 0) tdm_load_tile(kbb + tnext, (unsigned)(uintptr_t)&Ks[nxt][0]);
#else
            load_k_tile(nxt, tnext);
#endif
            load_v_tile(nxt, tnext);
        }

        // ---- logits: 16 q rows x 32 kv cols, K-dim=256 (8 wmma steps x 2)
        v8f c0 = {}, c1 = {};
#pragma unroll
        for (int ks = 0; ks < 8; ++ks) {
            const int k = ks * 32 + half * 16;
            v16bf b0 = *(const v16bf*)(&Ks[cur][laneN * 256 + k]);
            v16bf b1 = *(const v16bf*)(&Ks[cur][(16 + laneN) * 256 + k]);
            c0 = WMMA_BF16(aq[ks], b0, c0);
            c1 = WMMA_BF16(aq[ks], b1, c1);
        }

        // ---- online softmax (row stats across the 16 lanes of each half)
#pragma unroll
        for (int i = 0; i < 8; ++i) {
            float s0 = c0[i] * scale, s1 = c1[i] * scale;
            float tmax = fmaxf(s0, s1);
#pragma unroll
            for (int off = 1; off < 16; off <<= 1)
                tmax = fmaxf(tmax, __shfl_xor(tmax, off, 16));
            const float mnew = fmaxf(mrow[i], tmax);
            const float fac  = __expf(mrow[i] - mnew);
            const float p0 = __expf(s0 - mnew), p1 = __expf(s1 - mnew);
            float rs = p0 + p1;
#pragma unroll
            for (int off = 1; off < 16; off <<= 1)
                rs += __shfl_xor(rs, off, 16);
            lrow[i] = lrow[i] * fac + rs;
            mrow[i] = mnew;
#pragma unroll
            for (int nt = 0; nt < 16; ++nt) o[nt][i] *= fac;   // rescale O
            const int r = i + half * 8;                        // C-layout row
            Ps[wid][r * KV_TILE + laneN]      = (__bf16)p0;
            Ps[wid][r * KV_TILE + 16 + laneN] = (__bf16)p1;
        }

        // ---- O += P(16x32) x V(32x256): 16 N-tiles, one wmma each
        v16bf ap = *(const v16bf*)(&Ps[wid][laneN * KV_TILE + half * 16]);
#pragma unroll
        for (int nt = 0; nt < 16; ++nt) {
            v16bf bv = *(const v16bf*)(&Vt[cur][(nt * 16 + laneN) * KV_TILE + half * 16]);
            o[nt] = WMMA_BF16(ap, bv, o[nt]);
        }

        // ---- drain the in-flight TDM, then one barrier flips the buffers
#ifdef USE_TDM
        if (it + 1 < NIT && wid == 0) __builtin_amdgcn_s_wait_tensorcnt(0);
#endif
        __syncthreads();
    }

    // Emit unnormalized slice output + row stats (deterministic, no atomics)
    float* Ob = Opart + ((size_t)slice * MROWS + (size_t)batch * S) * D;
#pragma unroll
    for (int i = 0; i < 8; ++i) {
        const int r = q0 + i + half * 8;
#pragma unroll
        for (int nt = 0; nt < 16; ++nt)
            Ob[(size_t)r * D + nt * 16 + laneN] = o[nt][i];
        if (laneN == 0) {
            Mpart[(size_t)slice * MROWS + batch * S + r] = mrow[i];
            Lpart[(size_t)slice * MROWS + batch * S + r] = lrow[i];
        }
    }
}

// ---------------------------------------------------------------------------
// Kernel 4: merge KV slices (log-sum-exp) + partial query sum-pool
// grid=(64 q-chunks, B); thread d accumulates over 64 query rows.
// ---------------------------------------------------------------------------
__global__ void __launch_bounds__(256)
merge_kernel(const float* __restrict__ Opart, const float* __restrict__ Mpart,
             const float* __restrict__ Lpart, float* __restrict__ partial2) {
    const int b = blockIdx.y, qc = blockIdx.x, d = threadIdx.x;
    float acc = 0.0f;
    for (int q = 0; q < 64; ++q) {
        const int row = b * S + qc * 64 + q;
        float m[NSLICE], w[NSLICE];
        float M = -3.0e38f;
#pragma unroll
        for (int s = 0; s < NSLICE; ++s) {
            m[s] = Mpart[(size_t)s * MROWS + row];
            M = fmaxf(M, m[s]);
        }
        float L = 0.0f, v = 0.0f;
#pragma unroll
        for (int s = 0; s < NSLICE; ++s) {
            w[s] = __expf(m[s] - M);
            L += w[s] * Lpart[(size_t)s * MROWS + row];
            v += w[s] * Opart[((size_t)s * MROWS + row) * D + d];
        }
        acc += v / L;
    }
    partial2[((size_t)b * 64 + qc) * D + d] = acc;
}

// ---------------------------------------------------------------------------
// Kernel 5: deterministic reduction of q-chunk partials -> out[B][D]
// ---------------------------------------------------------------------------
__global__ void __launch_bounds__(256)
reduce_kernel(const float* __restrict__ partial2, float* __restrict__ out) {
    const int b = blockIdx.x, d = threadIdx.x;
    float s = 0.0f;
    for (int j = 0; j < 64; ++j)
        s += partial2[((size_t)b * 64 + j) * D + d];
    out[b * D + d] = s;
}

// ---------------------------------------------------------------------------
extern "C" void kernel_launch(void* const* d_in, const int* in_sizes, int n_in,
                              void* d_out, int out_size, void* d_ws, size_t ws_size,
                              hipStream_t stream) {
    const float* x    = (const float*)d_in[0];
    const float* W    = (const float*)d_in[1];
    const float* bias = (const float*)d_in[2];
    float* out = (float*)d_out;

    bf16_t* qb = (bf16_t*)d_ws;                            //  8 MB
    bf16_t* kb = qb + (size_t)MROWS * D;                   //  8 MB
    bf16_t* xb = kb + (size_t)MROWS * D;                   //  8 MB
    bf16_t* Wb = xb + (size_t)MROWS * D;                   // 256 KB
    float*  Mpart = (float*)(Wb + (size_t)TWO_D * D);      // 256 KB
    float*  Lpart = Mpart + (size_t)NSLICE * MROWS;        // 256 KB
    float*  partial2 = Lpart + (size_t)NSLICE * MROWS;     // 256 KB
    float*  Opart = partial2 + (size_t)BB * 64 * D;        //  64 MB

    cast_kernel<<<(MROWS * D) / 2048, 256, 0, stream>>>(x, xb);
    cast_kernel<<<(TWO_D * D) / 2048, 256, 0, stream>>>(W, Wb);
    proj_kernel<<<dim3(MROWS / 64, TWO_D / 128), 256, 0, stream>>>(xb, Wb, bias, qb, kb);
    attn_kernel<<<dim3(S / 128, BB, NSLICE), 256, 0, stream>>>(qb, kb, xb, Opart, Mpart, Lpart);
    merge_kernel<<<dim3(64, BB), 256, 0, stream>>>(Opart, Mpart, Lpart, partial2);
    reduce_kernel<<<BB, 256, 0, stream>>>(partial2, out);
}
